// simple_GAT_regression_154618822903
// MI455X (gfx1250) — matrix operand order repaired and checked by reading, not compiled, not run on hardware
//
#include <hip/hip_runtime.h>
#include <hip/hip_bf16.h>
#include <math.h>

#define NN 50000
#define EE 800000
#define CC 64
#define GG 256
#define ET (EE + NN)          // edges incl. self-loops
#define NEG_SLOPE 0.2f
#define BN_EPS 1e-5f
#define WT_STRIDE 72          // padded LDS stride (halves): 144B, 16B-aligned, bank-staggered

typedef __attribute__((ext_vector_type(16))) _Float16 v16h;
typedef __attribute__((ext_vector_type(8)))  float    v8f;

__device__ __forceinline__ void atomicMaxF(float* addr, float v) {
    // monotone-bit-pattern trick; init value is -inf so mixed signs are fine
    if (v >= 0.0f) atomicMax((int*)addr, __float_as_int(v));
    else           atomicMin((unsigned int*)addr, __float_as_uint(v));
}

// ---------------- layer 1: IN=1 -> h[n][c] = X[n] * W1[c] ----------------
__global__ __launch_bounds__(256) void k_layer1(const float* __restrict__ X,
                                                const float* __restrict__ W1,
                                                float* __restrict__ h) {
    int t = blockIdx.x * 256 + threadIdx.x;
    if (t >= NN * CC) return;
    int n = t >> 6, c = t & 63;
    h[t] = X[n] * W1[c];
}

// ---------------- layers 2..5: h = x @ W via v_wmma_f32_16x16x32_f16 ----------------
// one wave -> 16 rows x 64 cols; 8 waves/block -> 128 rows/block.
// W staged TRANSPOSED in LDS as f16 so each lane's B fragment is 16 contiguous
// halves -> 2x ds_load_b128 per fragment instead of 16x ds_load_u16.
__global__ __launch_bounds__(256) void k_gemm(const float* __restrict__ x,
                                              const float* __restrict__ W,
                                              float* __restrict__ h) {
    __shared__ _Float16 sWt[CC * WT_STRIDE];   // 9216 B of the WGP's 320 KB LDS
    #pragma unroll
    for (int i = 0; i < 16; ++i) {
        int j = threadIdx.x + (i << 8);        // coalesced global read of W[k][c]
        int k = j >> 6, c = j & 63;
        sWt[c * WT_STRIDE + k] = (_Float16)W[j];   // transposed store
    }
    __syncthreads();

    const int lane = threadIdx.x & 31;
    const int wave = threadIdx.x >> 5;
    const int row0 = (blockIdx.x * 8 + wave) * 16;
    if (row0 >= NN) return;                 // N % 16 == 0 -> full tiles only
    const int half = lane >> 4;             // 0: lanes 0-15, 1: lanes 16-31
    const int l15  = lane & 15;

    // A fragments (16x32 f16 each, covering K 0..31 and 32..63)
    const float* xr = x + (size_t)(row0 + l15) * CC;
    const int kb8 = half * 8;
    v16h a0, a1;
    #pragma unroll
    for (int i = 0; i < 8; ++i) {
        a0[i]     = (_Float16)xr[half * 16 + i];            // K = half*16 + i
        a0[i + 8] = (_Float16)xr[half * 16 + 8 + i];       // K = half*16 + 8 + i
        a1[i]     = (_Float16)xr[32 + half * 16 + i];
        a1[i + 8] = (_Float16)xr[32 + half * 16 + 8 + i];
    }

    const int kb16 = half * 16;
    #pragma unroll
    for (int j = 0; j < 4; ++j) {           // four 16-wide N tiles
        const int col = j * 16 + l15;
        // B fragments (32x16 f16): 16 contiguous halves per lane from sWt
        const _Float16* bp = sWt + col * WT_STRIDE + kb16;
        v16h b0 = *(const v16h*)(bp);        // K = half*16 + 0..15
        v16h b1 = *(const v16h*)(bp + 32);   // K = 32 + half*16 + 0..15
        v8f c = {};
        c = __builtin_amdgcn_wmma_f32_16x16x32_f16(false, a0, false, b0,
                                                   (short)0, c, false, false);
        c = __builtin_amdgcn_wmma_f32_16x16x32_f16(false, a1, false, b1,
                                                   (short)0, c, false, false);
        #pragma unroll
        for (int r = 0; r < 8; ++r)         // D layout: VGPR r -> M = r + 8*half
            h[(size_t)(row0 + half * 8 + r) * CC + col] = c[r];
    }
}

// ---------------- per-node attention logits: as[n]=h.a_s, ad[n]=h.a_d ----------------
__global__ __launch_bounds__(256) void k_alpha(const float* __restrict__ h,
                                               const float* __restrict__ a_s,
                                               const float* __restrict__ a_d,
                                               float* __restrict__ asv,
                                               float* __restrict__ adv) {
    const int lane = threadIdx.x & 31;
    const int node = blockIdx.x * 8 + (threadIdx.x >> 5);
    if (node >= NN) return;
    float2 hv = *(const float2*)(h + (size_t)node * CC + 2 * lane);
    float s = hv.x * a_s[2 * lane] + hv.y * a_s[2 * lane + 1];
    float d = hv.x * a_d[2 * lane] + hv.y * a_d[2 * lane + 1];
    #pragma unroll
    for (int off = 16; off > 0; off >>= 1) {
        s += __shfl_xor(s, off, 32);
        d += __shfl_xor(d, off, 32);
    }
    if (lane == 0) { asv[node] = s; adv[node] = d; }
}

// ---------------- init per-layer accumulators ----------------
__global__ __launch_bounds__(256) void k_init(float* __restrict__ mx,
                                              float* __restrict__ zs,
                                              float* __restrict__ obuf) {
    int t = blockIdx.x * 256 + threadIdx.x;
    if (t >= NN * CC) return;
    obuf[t] = 0.0f;
    if (t < NN) { mx[t] = -INFINITY; zs[t] = 0.0f; }
}

__device__ __forceinline__ void edge_sd(const int* __restrict__ ei, int e,
                                        int& s, int& d) {
    if (e < EE) { s = ei[e]; d = ei[EE + e]; }
    else        { s = e - EE; d = e - EE; }      // self loops
}

// pass 1: e = leaky_relu(as[src]+ad[dst]); segment max over dst
__global__ __launch_bounds__(256) void k_edge_logits(const int* __restrict__ ei,
                                                     const float* __restrict__ asv,
                                                     const float* __restrict__ adv,
                                                     float* __restrict__ ebuf,
                                                     float* __restrict__ mx) {
    int t = blockIdx.x * 256 + threadIdx.x;
    if (t >= ET) return;
    int s, d; edge_sd(ei, t, s, d);
    float e = asv[s] + adv[d];
    e = (e > 0.0f) ? e : NEG_SLOPE * e;
    ebuf[t] = e;
    atomicMaxF(&mx[d], e);
}

// pass 2: w = exp(e - m[dst]); segment sum over dst
__global__ __launch_bounds__(256) void k_edge_exp(const int* __restrict__ ei,
                                                  float* __restrict__ ebuf,
                                                  const float* __restrict__ mx,
                                                  float* __restrict__ zs) {
    int t = blockIdx.x * 256 + threadIdx.x;
    if (t >= ET) return;
    int s, d; edge_sd(ei, t, s, d);
    float w = __expf(ebuf[t] - mx[d]);
    ebuf[t] = w;
    atomicAdd(&zs[d], w);
}

// pass 3: out[dst] += h[src] * (w/z[dst]); one wave per edge, 2 channels/lane.
// edge index is wave-uniform: lane 0 does the scalar gathers, broadcast by shfl.
__global__ __launch_bounds__(256) void k_edge_agg(const int* __restrict__ ei,
                                                  const float* __restrict__ ebuf,
                                                  const float* __restrict__ zs,
                                                  const float* __restrict__ h,
                                                  float* __restrict__ obuf) {
    long long t = (long long)blockIdx.x * 256 + threadIdx.x;
    int e = (int)(t >> 5);
    if (e >= ET) return;
    int lane = (int)(t & 31);
    int s = 0, d = 0;
    float alpha = 0.0f;
    if (lane == 0) {
        edge_sd(ei, e, s, d);
        alpha = ebuf[e] / zs[d];
    }
    s     = __shfl(s, 0, 32);
    d     = __shfl(d, 0, 32);
    alpha = __shfl(alpha, 0, 32);
    float2 hv = *(const float2*)(h + (size_t)s * CC + 2 * lane);
    float* o = obuf + (size_t)d * CC + 2 * lane;
    atomicAdd(o,     hv.x * alpha);
    atomicAdd(o + 1, hv.y * alpha);
}

// ---------------- +bias, ReLU, BatchNorm (eval) -> next x ----------------
__global__ __launch_bounds__(256) void k_post(const float* __restrict__ obuf,
                                              const float* __restrict__ bias,
                                              const float* __restrict__ gam,
                                              const float* __restrict__ bet,
                                              const float* __restrict__ mu,
                                              const float* __restrict__ var,
                                              float* __restrict__ xbuf) {
    int t = blockIdx.x * 256 + threadIdx.x;
    if (t >= NN * CC) return;
    int c = t & 63;
    float v = obuf[t] + bias[c];
    v = fmaxf(v, 0.0f);
    v = (v - mu[c]) * (gam[c] * rsqrtf(var[c] + BN_EPS)) + bet[c];
    xbuf[t] = v;
}

// ---------------- pooling + head ----------------
__global__ __launch_bounds__(256) void k_pool_zero(float* __restrict__ pooled,
                                                   float* __restrict__ cnts) {
    int t = blockIdx.x * 256 + threadIdx.x;
    if (t >= GG * CC) return;
    pooled[t] = 0.0f;
    if (t < GG) cnts[t] = 0.0f;
}

__global__ __launch_bounds__(256) void k_pool_cnt(const int* __restrict__ batch,
                                                  float* __restrict__ cnts) {
    int t = blockIdx.x * 256 + threadIdx.x;
    if (t >= NN) return;
    atomicAdd(&cnts[batch[t]], 1.0f);
}

__global__ __launch_bounds__(256) void k_pool_sum(const int* __restrict__ batch,
                                                  const float* __restrict__ xbuf,
                                                  float* __restrict__ pooled) {
    int t = blockIdx.x * 256 + threadIdx.x;
    if (t >= NN * CC) return;
    int n = t >> 6, c = t & 63;
    atomicAdd(&pooled[(size_t)batch[n] * CC + c], xbuf[t]);
}

// finalize: pooled /= max(cnt,1); pred = pooled @ Wr + br. one wave per graph
__global__ __launch_bounds__(256) void k_pool_fin(float* __restrict__ pooled,
                                                  const float* __restrict__ cnts,
                                                  const float* __restrict__ Wr,
                                                  const float* __restrict__ br,
                                                  float* __restrict__ pred) {
    const int lane = threadIdx.x & 31;
    const int g = blockIdx.x * 8 + (threadIdx.x >> 5);
    if (g >= GG) return;
    float scale = 1.0f / fmaxf(cnts[g], 1.0f);
    float2* pp = (float2*)(pooled + (size_t)g * CC + 2 * lane);
    float2 p = *pp;
    p.x *= scale; p.y *= scale;
    *pp = p;
    float s = p.x * Wr[2 * lane] + p.y * Wr[2 * lane + 1];
    #pragma unroll
    for (int off = 16; off > 0; off >>= 1) s += __shfl_xor(s, off, 32);
    if (lane == 0) pred[g] = s + br[0];
}

extern "C" void kernel_launch(void* const* d_in, const int* in_sizes, int n_in,
                              void* d_out, int out_size, void* d_ws, size_t ws_size,
                              hipStream_t stream) {
    const float* X        = (const float*)d_in[0];
    const int*   ei       = (const int*)d_in[1];
    const int*   batch    = (const int*)d_in[2];
    const float* W1       = (const float*)d_in[3];
    const float* Ws       = (const float*)d_in[4];
    const float* att_src  = (const float*)d_in[5];
    const float* att_dst  = (const float*)d_in[6];
    const float* bias     = (const float*)d_in[7];
    const float* gam      = (const float*)d_in[8];
    const float* bet      = (const float*)d_in[9];
    const float* mu       = (const float*)d_in[10];
    const float* var      = (const float*)d_in[11];
    const float* Wr       = (const float*)d_in[12];
    const float* br       = (const float*)d_in[13];

    float* ws   = (float*)d_ws;
    float* xbuf = ws;                          // N*64
    float* hbuf = xbuf + (size_t)NN * CC;      // N*64
    float* obuf = hbuf + (size_t)NN * CC;      // N*64
    float* asv  = obuf + (size_t)NN * CC;      // N
    float* adv  = asv + NN;                    // N
    float* mx   = adv + NN;                    // N
    float* zs   = mx + NN;                     // N
    float* wb   = zs + NN;                     // E+N
    float* cnts = wb + ET;                     // G

    float* pred   = (float*)d_out;             // G
    float* pooled = pred + GG;                 // G*64

    dim3 blk(256);
    const int gridNC   = (NN * CC + 255) / 256;
    const int gridE    = (ET + 255) / 256;
    const int gridEA   = (int)(((long long)ET * 32 + 255) / 256);
    const int gridGemm = ((NN / 16) + 7) / 8;
    const int gridWvN  = (NN + 7) / 8;

    for (int L = 0; L < 5; ++L) {
        if (L == 0)
            k_layer1<<<gridNC, blk, 0, stream>>>(X, W1, hbuf);
        else
            k_gemm<<<gridGemm, blk, 0, stream>>>(xbuf, Ws + (size_t)(L - 1) * CC * CC, hbuf);
        k_alpha<<<gridWvN, blk, 0, stream>>>(hbuf, att_src + L * CC, att_dst + L * CC, asv, adv);
        k_init<<<gridNC, blk, 0, stream>>>(mx, zs, obuf);
        k_edge_logits<<<gridE, blk, 0, stream>>>(ei, asv, adv, wb, mx);
        k_edge_exp<<<gridE, blk, 0, stream>>>(ei, wb, mx, zs);
        k_edge_agg<<<gridEA, blk, 0, stream>>>(ei, wb, zs, hbuf, obuf);
        k_post<<<gridNC, blk, 0, stream>>>(obuf, bias + L * CC, gam + L * CC, bet + L * CC,
                                           mu + L * CC, var + L * CC, xbuf);
    }
    k_pool_zero<<<(GG * CC + 255) / 256, blk, 0, stream>>>(pooled, cnts);
    k_pool_cnt<<<(NN + 255) / 256, blk, 0, stream>>>(batch, cnts);
    k_pool_sum<<<gridNC, blk, 0, stream>>>(batch, xbuf, pooled);
    k_pool_fin<<<GG / 8, blk, 0, stream>>>(pooled, cnts, Wr, br, pred);
}
